// Attention_5274219839815
// MI455X (gfx1250) — matrix operand order repaired and checked by reading, not compiled, not run on hardware
//
#include <hip/hip_runtime.h>

typedef unsigned short u16;
typedef unsigned int   u32;
typedef unsigned long long u64;
typedef __attribute__((ext_vector_type(16))) __bf16 v16bf;
typedef __attribute__((ext_vector_type(8)))  float  v8f;
typedef __attribute__((ext_vector_type(2)))  __bf16 bf16x2;

// ---------------------------------------------------------------------------
// CDNA5 async global->LDS copy (ASYNCcnt-tracked), per CDNA5 ISA
// GVS mode: mem_addr = SADDR + VADDR + inst_offset; VDST = LDS byte address.
// LDS byte address = low 32 bits of the generic pointer (LDS aperture rule).
// ---------------------------------------------------------------------------
__device__ __forceinline__ void async_load_b128(u32 lds_addr, u64 gbase, u32 goff) {
    asm volatile("global_load_async_to_lds_b128 %0, %1, %2"
                 :: "v"(lds_addr), "v"(goff), "s"(gbase)
                 : "memory");
}
__device__ __forceinline__ void wait_asynccnt0() {
    asm volatile("s_wait_asynccnt 0x0" ::: "memory");
}
__device__ __forceinline__ u32 lds_addr_of(const void* p) {
    return (u32)(uintptr_t)p;
}

// ---------------------------------------------------------------------------
// WMMA fragment loaders (CDNA5 ISA 7.12.2 layouts, wave32)
// ---------------------------------------------------------------------------
// A-matrix 16x32 bf16 (MxK), src row-major, row stride ld elements.
__device__ __forceinline__ v16bf load_afrag(const __bf16* src, int ld, int lane) {
    const int row = lane & 15, half = lane >> 4;
    v16bf f;
#pragma unroll
    for (int v = 0; v < 8; ++v) {
        const int k = ((v >> 2) << 4) + (half << 3) + ((v & 3) << 1);
        bf16x2 p = *(const bf16x2*)(src + row * ld + k);
        f[2 * v]     = p.x;
        f[2 * v + 1] = p.y;
    }
    return f;
}

// B-matrix 32x16 bf16 (KxN). Source stored N-major: element (k,n) at srcT[n*ld + k].
__device__ __forceinline__ v16bf load_bfragT(const __bf16* srcT, int ld, int lane) {
    const int n = lane & 15, half = lane >> 4;
    v16bf f;
#pragma unroll
    for (int v = 0; v < 8; ++v) {
        const int k = ((v >> 2) << 4) + (half << 3) + ((v & 3) << 1);
        bf16x2 p = *(const bf16x2*)(srcT + n * ld + k);
        f[2 * v]     = p.x;
        f[2 * v + 1] = p.y;
    }
    return f;
}

#define WMMA_BF16(A, Bf, C) \
    __builtin_amdgcn_wmma_f32_16x16x32_bf16(false, (A), false, (Bf), (short)0, (C), false, false)

// ---------------------------------------------------------------------------
// Kernel 1: RMSNorm of tokens -> bf16 activations. One block per row of 1024.
// ---------------------------------------------------------------------------
__global__ __launch_bounds__(256)
void rmsnorm_kernel(const float* __restrict__ X, const float* __restrict__ W,
                    u16* __restrict__ Yb) {
    __shared__ float red[8];
    const int row = blockIdx.x, tid = threadIdx.x;
    const float4 xv = *(const float4*)(X + (size_t)row * 1024 + tid * 4);
    float ss = xv.x * xv.x + xv.y * xv.y + xv.z * xv.z + xv.w * xv.w;
#pragma unroll
    for (int m = 16; m >= 1; m >>= 1) ss += __shfl_xor(ss, m, 32);
    if ((tid & 31) == 0) red[tid >> 5] = ss;
    __syncthreads();
    float tot = 0.f;
#pragma unroll
    for (int i = 0; i < 8; ++i) tot += red[i];
    const float scale = rsqrtf(tot * (1.0f / 1024.0f) + 1e-6f);
    const float4 wv = *(const float4*)(W + tid * 4);
    __bf16* y = (__bf16*)(Yb + (size_t)row * 1024 + tid * 4);
    y[0] = (__bf16)(xv.x * scale * wv.x);
    y[1] = (__bf16)(xv.y * scale * wv.y);
    y[2] = (__bf16)(xv.z * scale * wv.z);
    y[3] = (__bf16)(xv.w * scale * wv.w);
}

// ---------------------------------------------------------------------------
// Kernel 2: fp32 -> bf16 conversion (4 elems/thread, exact grid)
// ---------------------------------------------------------------------------
__global__ __launch_bounds__(256)
void cvt_bf16_kernel(const float* __restrict__ in, u16* __restrict__ outb, int n) {
    const int idx = (blockIdx.x * 256 + threadIdx.x) * 4;
    if (idx + 3 < n) {
        float4 v = *(const float4*)(in + idx);
        __bf16* o = (__bf16*)(outb + idx);
        o[0] = (__bf16)v.x; o[1] = (__bf16)v.y; o[2] = (__bf16)v.z; o[3] = (__bf16)v.w;
    }
}

// ---------------------------------------------------------------------------
// Kernel 3: bf16 GEMM  C[f32] = A[MxK] * B[KxN], 128x128x32 tiles,
// double-buffered LDS; A tile staged via async-to-LDS, B tile via regs
// (global loads for k+1 issued before compute of k).
// ---------------------------------------------------------------------------
#define TBM 128
#define TBN 128
#define TBK 32

__global__ __launch_bounds__(256)
void gemm_bf16_kernel(const u16* __restrict__ Ab, const u16* __restrict__ Bb,
                      float* __restrict__ C, int M, int K, int N) {
    (void)M;
    __shared__ u16 As[2][TBM * TBK];    // [m][k] row-major
    __shared__ u16 BsT[2][TBN * TBK];   // [n][k] transposed
    const int tid = threadIdx.x, lane = tid & 31, wave = tid >> 5;
    const int m0 = blockIdx.y * TBM, n0 = blockIdx.x * TBN;
    const __bf16* Bg = (const __bf16*)Bb;

    // A staging: thread t -> row ar = t>>1, cols [ac, ac+16): two b128 async copies
    const int ar = tid >> 1, ac = (tid & 1) << 4;
    const u64 abase = (u64)(uintptr_t)Ab;
    const u32 aoff0 = (u32)(((m0 + ar) * K + ac) * 2);   // byte offset at k0 = 0
    // B staging: row bk = t>>3 (32 rows), cols [bc0, bc0+16)
    const int bk = tid >> 3, bc0 = (tid & 7) << 4;

    v8f acc[8];
#pragma unroll
    for (int i = 0; i < 8; ++i) acc[i] = {};

    uint4 br0 = {}, br1 = {};
    {   // prologue: stage tile 0 into buffer 0
        const u32 lds_a = lds_addr_of(&As[0][ar * TBK + ac]);
        async_load_b128(lds_a,      abase, aoff0);
        async_load_b128(lds_a + 16, abase, aoff0 + 16);
        const uint4* g = (const uint4*)(Bg + (size_t)bk * N + n0 + bc0);
        br0 = g[0]; br1 = g[1];
        union { uint4 v[2]; u16 h[16]; } u; u.v[0] = br0; u.v[1] = br1;
#pragma unroll
        for (int j = 0; j < 16; ++j) BsT[0][(bc0 + j) * TBK + bk] = u.h[j];
    }
    wait_asynccnt0();
    __syncthreads();

    const int nk = K / TBK;
    for (int kt = 0; kt < nk; ++kt) {
        const int cur = kt & 1, nxt = cur ^ 1;
        const bool more = (kt + 1 < nk);
        if (more) {  // start next tile's fills before computing on current
            const u32 lds_a = lds_addr_of(&As[nxt][ar * TBK + ac]);
            const u32 go = aoff0 + (u32)((kt + 1) * TBK * 2);
            async_load_b128(lds_a,      abase, go);
            async_load_b128(lds_a + 16, abase, go + 16);
            const uint4* g = (const uint4*)(Bg + (size_t)((kt + 1) * TBK + bk) * N + n0 + bc0);
            br0 = g[0]; br1 = g[1];
        }
        // compute on current buffers; rotate B-fragment prefetch one step ahead
        const v16bf af = load_afrag((const __bf16*)As[cur] + (wave * 16) * TBK, TBK, lane);
        v16bf bcur = load_bfragT((const __bf16*)BsT[cur], TBK, lane);
#pragma unroll
        for (int ns = 0; ns < 8; ++ns) {
            v16bf bnxt = bcur;
            if (ns < 7)
                bnxt = load_bfragT((const __bf16*)BsT[cur] + ((ns + 1) * 16) * TBK, TBK, lane);
            acc[ns] = WMMA_BF16(af, bcur, acc[ns]);
            bcur = bnxt;
        }
        if (more) {  // transpose-store next B tile after compute
            union { uint4 v[2]; u16 h[16]; } u; u.v[0] = br0; u.v[1] = br1;
#pragma unroll
            for (int j = 0; j < 16; ++j) BsT[nxt][(bc0 + j) * TBK + bk] = u.h[j];
        }
        wait_asynccnt0();
        __syncthreads();
    }

    const int half = lane >> 4, nn = lane & 15;
#pragma unroll
    for (int ns = 0; ns < 8; ++ns)
#pragma unroll
        for (int r = 0; r < 8; ++r)
            C[(size_t)(m0 + wave * 16 + r + 8 * half) * N + n0 + ns * 16 + nn] = acc[ns][r];
}

// ---------------------------------------------------------------------------
// Kernel 4: per-head L2-norm (q,k) + relayout to [B*H][N][64] bf16, v copied.
// ---------------------------------------------------------------------------
__global__ __launch_bounds__(256)
void qkv_prep_kernel(const float* __restrict__ Qf, const float* __restrict__ KVf,
                     const float* __restrict__ qg, const float* __restrict__ kg,
                     u16* __restrict__ Qn, u16* __restrict__ Kn, u16* __restrict__ Vn) {
    const int tid = threadIdx.x, lane = tid & 31, wave = tid >> 5;
    const int rid = blockIdx.x * 8 + wave;           // (b*16+h)*2048 + n
    const int b = rid >> 15, h = (rid >> 11) & 15, n = rid & 2047;
    const int d0 = lane * 2;
    const size_t qoff = ((size_t)(b * 2048 + n)) * 1024 + h * 64 + d0;
    const size_t koff = ((size_t)(b * 2048 + n)) * 2048 + h * 64 + d0;
    const float2 q = *(const float2*)(Qf + qoff);
    const float2 k = *(const float2*)(KVf + koff);
    const float2 v = *(const float2*)(KVf + koff + 1024);
    float qss = q.x * q.x + q.y * q.y;
    float kss = k.x * k.x + k.y * k.y;
#pragma unroll
    for (int m = 16; m >= 1; m >>= 1) {
        qss += __shfl_xor(qss, m, 32);
        kss += __shfl_xor(kss, m, 32);
    }
    const float qs = 8.0f / fmaxf(sqrtf(qss), 1e-12f);   // MHN_SCALE = sqrt(64)
    const float ks = 8.0f / fmaxf(sqrtf(kss), 1e-12f);
    const float2 qgv = *(const float2*)(qg + h * 64 + d0);
    const float2 kgv = *(const float2*)(kg + h * 64 + d0);
    const size_t oo = (size_t)rid * 64 + d0;
    __bf16* qo = (__bf16*)Qn; __bf16* ko = (__bf16*)Kn; __bf16* vo = (__bf16*)Vn;
    qo[oo]     = (__bf16)(q.x * qs * (qgv.x + 1.0f));
    qo[oo + 1] = (__bf16)(q.y * qs * (qgv.y + 1.0f));
    ko[oo]     = (__bf16)(k.x * ks * (kgv.x + 1.0f));
    ko[oo + 1] = (__bf16)(k.y * ks * (kgv.y + 1.0f));
    vo[oo]     = (__bf16)v.x;
    vo[oo + 1] = (__bf16)v.y;
}

// ---------------------------------------------------------------------------
// Kernel 5: streaming attention per (b,h), double-buffered K/V tiles.
// Softcap bounds logits to +/-6.25 -> streaming softmax without running max.
// ---------------------------------------------------------------------------
__global__ __launch_bounds__(128)
void attn_kernel(const u16* __restrict__ Qb, const u16* __restrict__ Kb,
                 const u16* __restrict__ Vb, u16* __restrict__ Ob) {
    __shared__ u16 Kt[2][64 * 64];    // [key][d]
    __shared__ u16 VtT[2][64 * 64];   // [d][key]
    __shared__ u16 Ps[4][16 * 64];    // per-wave P staging (also Q staging)
    const int tid = threadIdx.x, lane = tid & 31, wave = tid >> 5;
    const int bh = blockIdx.y, qb = blockIdx.x;
    const int half = lane >> 4, nn = lane & 15;
    const u64 qbase = (u64)(uintptr_t)Qb;
    const u64 kbase = (u64)(uintptr_t)Kb;
    const __bf16* Vn = (const __bf16*)Vb;

    {   // stage this wave's 16x64 Q block (async, wave-local -> no barrier needed)
        const u32 lds_q = lds_addr_of(&Ps[wave][0]) + (u32)(lane * 16);
        const u32 go = (u32)(((bh * 2048 + qb * 64 + wave * 16) * 64 + lane * 8) * 2);
#pragma unroll
        for (int j = 0; j < 4; ++j)
            async_load_b128(lds_q + j * 512, qbase, go + j * 512);
        wait_asynccnt0();
    }
    const v16bf qa0 = load_afrag((const __bf16*)Ps[wave], 64, lane);
    const v16bf qa1 = load_afrag((const __bf16*)Ps[wave] + 32, 64, lane);

    v8f o[4];
#pragma unroll
    for (int i = 0; i < 4; ++i) o[i] = {};
    float lsum[8];
#pragma unroll
    for (int r = 0; r < 8; ++r) lsum[r] = 0.f;

    // V staging: key = tid>>1, 32 contiguous d at dd0 = (tid&1)*32
    const int vkey = tid >> 1, dd0 = (tid & 1) << 5;
    uint4 vr[4];
    {   // prologue: stage K0 (async) and V0 (regs -> transposed LDS)
        const u32 ldsk = lds_addr_of(&Kt[0][0]) + (u32)(tid * 16);
        const u32 gk = (u32)(((bh * 2048) * 64 + tid * 8) * 2);
#pragma unroll
        for (int j = 0; j < 4; ++j)
            async_load_b128(ldsk + j * 2048, kbase, gk + j * 2048);
        const uint4* gv = (const uint4*)(Vn + ((size_t)bh * 2048) * 64 + vkey * 64 + dd0);
#pragma unroll
        for (int j = 0; j < 4; ++j) vr[j] = gv[j];
        union { uint4 v[4]; u16 h[32]; } u;
#pragma unroll
        for (int j = 0; j < 4; ++j) u.v[j] = vr[j];
#pragma unroll
        for (int j = 0; j < 32; ++j) VtT[0][(dd0 + j) * 64 + vkey] = u.h[j];
    }
    wait_asynccnt0();
    __syncthreads();

    for (int kb = 0; kb < 32; ++kb) {
        const int cur = kb & 1, nxt = cur ^ 1;
        const bool more = (kb + 1 < 32);
        if (more) {  // start next K (async) + next V (regs) before compute
            const u32 ldsk = lds_addr_of(&Kt[nxt][0]) + (u32)(tid * 16);
            const u32 gk = (u32)(((bh * 2048 + (kb + 1) * 64) * 64 + tid * 8) * 2);
#pragma unroll
            for (int j = 0; j < 4; ++j)
                async_load_b128(ldsk + j * 2048, kbase, gk + j * 2048);
            const uint4* gv =
                (const uint4*)(Vn + ((size_t)bh * 2048 + (kb + 1) * 64) * 64 + vkey * 64 + dd0);
#pragma unroll
            for (int j = 0; j < 4; ++j) vr[j] = gv[j];
        }

        // S = Q K^T -> softcap -> exp -> P (bf16 staged per-wave)
        float p[4][8];
#pragma unroll
        for (int ns = 0; ns < 4; ++ns) {
            v8f s = {};
            const v16bf b0 = load_bfragT((const __bf16*)Kt[cur] + (ns * 16) * 64, 64, lane);
            const v16bf b1 = load_bfragT((const __bf16*)Kt[cur] + (ns * 16) * 64 + 32, 64, lane);
            s = WMMA_BF16(qa0, b0, s);
            s = WMMA_BF16(qa1, b1, s);
#pragma unroll
            for (int r = 0; r < 8; ++r) {
                const float val = tanhf(s[r] * 0.02f) * 6.25f;  // tanh(x/50)*50*scale
                const float e = __expf(val);
                p[ns][r] = e;
                lsum[r] += e;
            }
        }
        __bf16* pw = (__bf16*)Ps[wave];
#pragma unroll
        for (int ns = 0; ns < 4; ++ns)
#pragma unroll
            for (int r = 0; r < 8; ++r)
                pw[(r + 8 * half) * 64 + ns * 16 + nn] = (__bf16)p[ns][r];
        // wave-internal LDS ops are in-order: read back immediately
        const v16bf pa0 = load_afrag((const __bf16*)Ps[wave], 64, lane);
        const v16bf pa1 = load_afrag((const __bf16*)Ps[wave] + 32, 64, lane);
#pragma unroll
        for (int nd = 0; nd < 4; ++nd) {
            const v16bf b0 = load_bfragT((const __bf16*)VtT[cur] + (nd * 16) * 64, 64, lane);
            const v16bf b1 = load_bfragT((const __bf16*)VtT[cur] + (nd * 16) * 64 + 32, 64, lane);
            o[nd] = WMMA_BF16(pa0, b0, o[nd]);
            o[nd] = WMMA_BF16(pa1, b1, o[nd]);
        }

        if (more) {  // transpose-store next V tile after compute
            union { uint4 v[4]; u16 h[32]; } u;
#pragma unroll
            for (int j = 0; j < 4; ++j) u.v[j] = vr[j];
#pragma unroll
            for (int j = 0; j < 32; ++j) VtT[nxt][(dd0 + j) * 64 + vkey] = u.h[j];
        }
        wait_asynccnt0();
        __syncthreads();
    }

    // finalize: row-sum across the 16 lanes holding each row, divide, store merged
#pragma unroll
    for (int r = 0; r < 8; ++r) {
        float v = lsum[r];
        v += __shfl_xor(v, 1, 32);
        v += __shfl_xor(v, 2, 32);
        v += __shfl_xor(v, 4, 32);
        v += __shfl_xor(v, 8, 32);
        lsum[r] = 1.0f / v;
    }
    const int b = bh >> 4, h = bh & 15;
    __bf16* om = (__bf16*)Ob;
#pragma unroll
    for (int nd = 0; nd < 4; ++nd)
#pragma unroll
        for (int r = 0; r < 8; ++r) {
            const int nq = qb * 64 + wave * 16 + r + 8 * half;
            om[((size_t)(b * 2048 + nq)) * 1024 + h * 64 + nd * 16 + nn] =
                (__bf16)(o[nd][r] * lsum[r]);
        }
}

// ---------------------------------------------------------------------------
// Host launcher
// ---------------------------------------------------------------------------
extern "C" void kernel_launch(void* const* d_in, const int* in_sizes, int n_in,
                              void* d_out, int out_size, void* d_ws, size_t ws_size,
                              hipStream_t stream) {
    (void)in_sizes; (void)n_in; (void)out_size; (void)ws_size;
    const float* tokens  = (const float*)d_in[0];
    const float* norm_w  = (const float*)d_in[1];
    const float* w_q     = (const float*)d_in[2];
    const float* w_kv    = (const float*)d_in[3];
    const float* w_out   = (const float*)d_in[4];
    const float* q_gamma = (const float*)d_in[5];
    const float* k_gamma = (const float*)d_in[6];
    float* out = (float*)d_out;

    char* ws = (char*)d_ws;
    const size_t MBy = 1024ull * 1024ull;
    u16*   x_bf    = (u16*)(ws);              //  8 MB  bf16 [4096,1024]
    u16*   wq_bf   = (u16*)(ws + 8 * MBy);    //  2 MB
    u16*   wkv_bf  = (u16*)(ws + 10 * MBy);   //  4 MB
    u16*   wout_bf = (u16*)(ws + 14 * MBy);   //  2 MB
    float* qf      = (float*)(ws + 16 * MBy); // 16 MB  f32 [4096,1024]
    float* kvf     = (float*)(ws + 32 * MBy); // 32 MB  f32 [4096,2048]
    u16*   qn      = (u16*)(ws + 64 * MBy);   //  8 MB  bf16 [32,2048,64]
    u16*   kn      = (u16*)(ws + 72 * MBy);   //  8 MB
    u16*   vn      = (u16*)(ws + 80 * MBy);   //  8 MB
    u16*   om      = (u16*)(ws + 16 * MBy);   //  8 MB  aliases qf (dead after prep)

    rmsnorm_kernel<<<4096, 256, 0, stream>>>(tokens, norm_w, x_bf);
    cvt_bf16_kernel<<<1024, 256, 0, stream>>>(w_q,   wq_bf,   1024 * 1024);
    cvt_bf16_kernel<<<2048, 256, 0, stream>>>(w_kv,  wkv_bf,  1024 * 2048);
    cvt_bf16_kernel<<<1024, 256, 0, stream>>>(w_out, wout_bf, 1024 * 1024);

    gemm_bf16_kernel<<<dim3(8, 32),  256, 0, stream>>>(x_bf, wq_bf,  qf,  4096, 1024, 1024);
    gemm_bf16_kernel<<<dim3(16, 32), 256, 0, stream>>>(x_bf, wkv_bf, kvf, 4096, 1024, 2048);

    qkv_prep_kernel<<<8192, 256, 0, stream>>>(qf, kvf, q_gamma, k_gamma, qn, kn, vn);

    attn_kernel<<<dim3(32, 32), 128, 0, stream>>>(qn, kn, vn, om);

    gemm_bf16_kernel<<<dim3(8, 32), 256, 0, stream>>>(om, wout_bf, out, 4096, 1024, 1024);
}
